// CrossAttnLayer_27427661152527
// MI455X (gfx1250) — matrix operand order repaired and checked by reading, not compiled, not run on hardware
//
#include <hip/hip_runtime.h>
#include <hip/hip_bf16.h>
#include <math.h>

typedef __attribute__((ext_vector_type(16))) __bf16 v16bf;
typedef __attribute__((ext_vector_type(8)))  __bf16 v8bf;
typedef __attribute__((ext_vector_type(4)))  __bf16 v4bf;
typedef __attribute__((ext_vector_type(8)))  float  v8f;
typedef __attribute__((ext_vector_type(2)))  int    v2i;

// Build a 16-element fragment from two contiguous 8-element halves.
__device__ __forceinline__ v16bf cat8(v8bf lo, v8bf hi) {
    union { struct { v8bf lo, hi; } s; v16bf v; } u;
    u.s.lo = lo; u.s.hi = hi;
    return u.v;
}

// ---------------------------------------------------------------------------
// CDNA5 async global->LDS copy (8 bytes per lane).
// gfx1250 builtin: (v2i addrspace(1)*, v2i addrspace(3)*, imm offset, imm cpol)
// ---------------------------------------------------------------------------
__device__ __forceinline__ void cp_async_b64(const __bf16* g, __bf16* l) {
#if __has_builtin(__builtin_amdgcn_global_load_async_to_lds_b64)
    __builtin_amdgcn_global_load_async_to_lds_b64(
        (v2i __attribute__((address_space(1)))*)(g),
        (v2i __attribute__((address_space(3)))*)(l), 0, 0);
#else
    *(uint2*)l = *(const uint2*)g;
#endif
}
__device__ __forceinline__ void cp_async_wait() {
#if __has_builtin(__builtin_amdgcn_global_load_async_to_lds_b64)
 #if __has_builtin(__builtin_amdgcn_s_wait_asynccnt)
    __builtin_amdgcn_s_wait_asynccnt(0);
 #else
    asm volatile("s_wait_asynccnt 0" ::: "memory");
 #endif
#endif
}

// ---------------------------------------------------------------------------
// Fragment layouts (ISA 7.12.2, 16-bit, wave32):
//  A 16x32: lane(0-15)=row M, elems 0..7 -> K=hi*8.., elems 8..15 -> K=16+hi*8..
//  B 32x16: lane&15 = col N, K order as A (elems 0..7 -> K=hi*8.., 8..15 -> K=16+hi*8..)
//  C/D:     lane&15 = N, VGPR i -> M = i + (lane>>4)*8.
// ---------------------------------------------------------------------------

// Epilogue modes (compile-time):
//  0: bf16 out                2: bf16 out, transposed per head (Vt[B,H,64,1024])
//  1: bf16 out + ReLU         4: f32 out + residual add
template <int MODE>
__global__ __launch_bounds__(256) void gemm_bf16_wmma(
    const __bf16* __restrict__ A, const __bf16* __restrict__ W,
    const float* __restrict__ bias, const float* __restrict__ res,
    float* __restrict__ Cf, __bf16* __restrict__ Cb,
    int M, int N, int K)
{
    __shared__ __bf16 Asb[32][40];    // 32x32 + pad (rows 16B-aligned)
    __shared__ __bf16 Bst[128][40];   // transposed [n][k] + pad

    const int tid  = threadIdx.x;
    const int wave = tid >> 5;
    const int lane = tid & 31;
    const int nl   = lane & 15;
    const int hi   = lane >> 4;
    const int m0   = blockIdx.y * 32;
    const int n0   = blockIdx.x * 128;
    const int nW   = wave * 16;

    v8f acc0 = {}, acc1 = {};
    for (int k0 = 0; k0 < K; k0 += 32) {
        // ---- stage A tile 32x32 bf16: one async 8B chunk per thread ----
        {
            int r = tid >> 3, co = (tid & 7) * 4;
            cp_async_b64(&A[(long)(m0 + r) * K + k0 + co], &Asb[r][co]);
        }
        // ---- stage B tile 32x128 bf16, transposed into Bst[n][k] ----
        #pragma unroll
        for (int t = 0; t < 2; ++t) {
            int idx = t * 256 + tid;
            int r = idx >> 4, c8 = (idx & 15) * 8;
            v8bf w8 = *(const v8bf*)&W[(long)(k0 + r) * N + n0 + c8];
            #pragma unroll
            for (int j = 0; j < 8; ++j) Bst[c8 + j][r] = w8[j];
        }
        cp_async_wait();
        __syncthreads();

        // ---- fragments: contiguous 16B LDS loads ----
        v16bf bfv = cat8(*(const v8bf*)&Bst[nW + nl][hi * 8],
                         *(const v8bf*)&Bst[nW + nl][16 + hi * 8]);
        v16bf a0  = cat8(*(const v8bf*)&Asb[nl][hi * 8],
                         *(const v8bf*)&Asb[nl][16 + hi * 8]);
        v16bf a1  = cat8(*(const v8bf*)&Asb[16 + nl][hi * 8],
                         *(const v8bf*)&Asb[16 + nl][16 + hi * 8]);
        acc0 = __builtin_amdgcn_wmma_f32_16x16x32_bf16(
                   false, a0, false, bfv, (short)0, acc0, false, false);
        acc1 = __builtin_amdgcn_wmma_f32_16x16x32_bf16(
                   false, a1, false, bfv, (short)0, acc1, false, false);
        __syncthreads();
    }

    const int n = n0 + nW + nl;
    const float bv = bias[n];
    #pragma unroll
    for (int t = 0; t < 2; ++t) {
        const v8f& acc = t ? acc1 : acc0;
        #pragma unroll
        for (int i = 0; i < 8; ++i) {
            const int m = m0 + t * 16 + i + hi * 8;
            float v = acc[i] + bv;
            if constexpr (MODE == 4) {
                v += res[(long)m * N + n];
                Cf[(long)m * N + n] = v;
            } else if constexpr (MODE == 2) {
                int b = m >> 10, s = m & 1023;      // S = 1024
                int h = n >> 6,  d = n & 63;        // dk = 64
                Cb[(long)(((b << 3) + h) * 64 + d) * 1024 + s] = (__bf16)v;
            } else {
                if constexpr (MODE == 1) v = v > 0.0f ? v : 0.0f;
                Cb[(long)m * N + n] = (__bf16)v;
            }
        }
    }
}

// ---------------------------------------------------------------------------
// Attention core for one (b, h, 16-row q tile).
// Q,K: bf16 [B*S, D] head-sliced. Vt: bf16 [B,H,64,Sk] (pre-transposed).
// mask int32 [B,Sq,Sk]; masked scores filled with 1e-9 (reference EPS_MASK).
// O: bf16 [B*Sq, D].
// ---------------------------------------------------------------------------
__global__ __launch_bounds__(256) void attn_wmma(
    const __bf16* __restrict__ Q, const __bf16* __restrict__ Km,
    const __bf16* __restrict__ Vt, const int* __restrict__ mask,
    __bf16* __restrict__ O, int Bn, int Hn, int Sq, int Sk, float scale)
{
    __shared__ __bf16 P[16][1024];     // 32 KB: scores -> probabilities
    __shared__ float  red[16][16];
    __shared__ float  rowmax[16], rowsum[16];
    __shared__ float  obuf[8][16][16]; // split-K partials

    const int tid  = threadIdx.x;
    const int wave = tid >> 5;
    const int lane = tid & 31;
    const int nl   = lane & 15;
    const int hi   = lane >> 4;

    const int nq = Sq >> 4;
    const int qt = blockIdx.x % nq;
    const int bh = blockIdx.x / nq;
    const int h  = bh % Hn;
    const int b  = bh / Hn;
    const int q0 = qt << 4;
    const int D  = Hn * 64;

    const __bf16* Qb  = Q  + (long)(b * Sq) * D + h * 64;
    const __bf16* Kb  = Km + (long)(b * Sk) * D + h * 64;
    const __bf16* Vtb = Vt + (long)(b * Hn + h) * 64 * Sk;
    const int*    Mb  = mask + (long)b * Sq * Sk;

    // ---- Phase 1: S = (Q K^T) * scale, mask-fill, bf16 scores in LDS ----
    const __bf16* qrow = Qb + (long)(q0 + nl) * D;
    for (int kt = wave; kt < (Sk >> 4); kt += 8) {
        v8f acc = {};
        const __bf16* krow = Kb + (long)(kt * 16 + nl) * D;
        #pragma unroll
        for (int kd = 0; kd < 64; kd += 32) {
            v16bf af = cat8(*(const v8bf*)&qrow[kd + hi * 8],
                            *(const v8bf*)&qrow[kd + 16 + hi * 8]);
            v16bf bf = cat8(*(const v8bf*)&krow[kd + hi * 8],
                            *(const v8bf*)&krow[kd + 16 + hi * 8]);
            acc = __builtin_amdgcn_wmma_f32_16x16x32_bf16(
                      false, af, false, bf, (short)0, acc, false, false);
        }
        #pragma unroll
        for (int i = 0; i < 8; ++i) {
            int m = i + hi * 8;
            int c = kt * 16 + nl;
            float s = acc[i] * scale;
            if (Mb[(long)(q0 + m) * Sk + c] == 0) s = 1e-9f;
            P[m][c] = (__bf16)s;
        }
    }
    __syncthreads();

    // ---- Phase 2: row softmax; each thread owns a contiguous 64-col chunk ----
    {
        const int r = tid >> 4, sub = tid & 15;
        const int c0 = sub * (Sk >> 4);          // 64-wide chunk
        const int cw = Sk >> 4;

        float mx = -3.0e38f;
        for (int c = c0; c < c0 + cw; c += 8) {
            v8bf v = *(const v8bf*)&P[r][c];
            #pragma unroll
            for (int j = 0; j < 8; ++j) mx = fmaxf(mx, (float)v[j]);
        }
        red[r][sub] = mx;
        __syncthreads();
        if (sub == 0) {
            float m2 = red[r][0];
            for (int j = 1; j < 16; ++j) m2 = fmaxf(m2, red[r][j]);
            rowmax[r] = m2;
        }
        __syncthreads();
        float rmx = rowmax[r], sm = 0.0f;
        for (int c = c0; c < c0 + cw; c += 8) {
            v8bf v = *(const v8bf*)&P[r][c];
            v8bf o;
            #pragma unroll
            for (int j = 0; j < 8; ++j) {
                float p = __expf((float)v[j] - rmx);
                o[j] = (__bf16)p;
                sm += p;
            }
            *(v8bf*)&P[r][c] = o;
        }
        red[r][sub] = sm;
        __syncthreads();
        if (sub == 0) {
            float s2 = 0.0f;
            for (int j = 0; j < 16; ++j) s2 += red[r][j];
            rowsum[r] = s2;
        }
        __syncthreads();
        float inv = 1.0f / rowsum[r];
        for (int c = c0; c < c0 + cw; c += 8) {
            v8bf v = *(const v8bf*)&P[r][c];
            v8bf o;
            #pragma unroll
            for (int j = 0; j < 8; ++j) o[j] = (__bf16)((float)v[j] * inv);
            *(v8bf*)&P[r][c] = o;
        }
    }
    __syncthreads();

    // ---- Phase 3: O = P @ V via Vt; split-K across wave pairs ----
    {
        const int nt = wave & 3;                 // dk segment
        const int kh = wave >> 2;                // K half
        const int kbeg = kh * (Sk >> 1);
        const int kend = kbeg + (Sk >> 1);
        const __bf16* vrow = Vtb + (long)(nt * 16 + nl) * Sk;
        v8f acc = {};
        for (int k0 = kbeg; k0 < kend; k0 += 32) {
            v16bf af = cat8(*(const v8bf*)&P[nl][k0 + hi * 8],
                            *(const v8bf*)&P[nl][k0 + 16 + hi * 8]);
            v16bf bf = cat8(*(const v8bf*)&vrow[k0 + hi * 8],
                            *(const v8bf*)&vrow[k0 + 16 + hi * 8]);
            acc = __builtin_amdgcn_wmma_f32_16x16x32_bf16(
                      false, af, false, bf, (short)0, acc, false, false);
        }
        #pragma unroll
        for (int i = 0; i < 8; ++i)
            obuf[wave][i + hi * 8][nl] = acc[i];
    }
    __syncthreads();

    // ---- combine split-K halves; one 16B store per thread (tid < 128) ----
    __bf16* Ob = O + (long)(b * Sq) * D + h * 64;
    if (tid < 128) {
        int m  = tid >> 3;           // 16 rows
        int c8 = (tid & 7) * 8;      // 8 chunks of 8 cols
        int nt = c8 >> 4;            // dk segment (c8..c8+7 within one segment)
        int n8 = c8 & 15;            // 0 or 8
        v8bf o;
        #pragma unroll
        for (int j = 0; j < 8; ++j)
            o[j] = (__bf16)(obuf[nt][m][n8 + j] + obuf[nt + 4][m][n8 + j]);
        *(v8bf*)&Ob[(long)(q0 + m) * D + c8] = o;
    }
}

// ---------------------------------------------------------------------------
// Elementwise helpers
// ---------------------------------------------------------------------------
__global__ __launch_bounds__(256) void cvt_bf16_kernel(
    const float* __restrict__ x, __bf16* __restrict__ y, int n4)
{
    int i = blockIdx.x * 256 + threadIdx.x;
    if (i < n4) {
        float4 v = ((const float4*)x)[i];
        v4bf o; o[0] = (__bf16)v.x; o[1] = (__bf16)v.y;
        o[2] = (__bf16)v.z; o[3] = (__bf16)v.w;
        *(v4bf*)&y[i * 4] = o;
    }
}

__global__ __launch_bounds__(256) void add2_bf16_kernel(
    const float* __restrict__ a, const float* __restrict__ b,
    __bf16* __restrict__ c, int n4)
{
    int i = blockIdx.x * 256 + threadIdx.x;
    if (i < n4) {
        float4 va = ((const float4*)a)[i];
        float4 vb = ((const float4*)b)[i];
        v4bf o; o[0] = (__bf16)(va.x + vb.x); o[1] = (__bf16)(va.y + vb.y);
        o[2] = (__bf16)(va.z + vb.z); o[3] = (__bf16)(va.w + vb.w);
        *(v4bf*)&c[i * 4] = o;
    }
}

__global__ __launch_bounds__(256) void ln_kernel(
    const float* __restrict__ x, const float* __restrict__ g,
    const float* __restrict__ bta, float* __restrict__ y,
    __bf16* __restrict__ yb, int D)
{
    __shared__ float sred[256];
    __shared__ float smu, srstd;
    const long row = blockIdx.x;
    const float* xr = x + row * D;
    const int tid = threadIdx.x;

    float s = 0.0f;
    for (int c = tid; c < D; c += 256) s += xr[c];
    sred[tid] = s; __syncthreads();
    for (int off = 128; off > 0; off >>= 1) {
        if (tid < off) sred[tid] += sred[tid + off];
        __syncthreads();
    }
    if (tid == 0) smu = sred[0] / (float)D;
    __syncthreads();
    const float mu = smu;

    float v = 0.0f;
    for (int c = tid; c < D; c += 256) { float d = xr[c] - mu; v += d * d; }
    sred[tid] = v; __syncthreads();
    for (int off = 128; off > 0; off >>= 1) {
        if (tid < off) sred[tid] += sred[tid + off];
        __syncthreads();
    }
    if (tid == 0) srstd = rsqrtf(sred[0] / (float)D + 1e-5f);
    __syncthreads();
    const float rstd = srstd;

    for (int c = tid; c < D; c += 256) {
        float o = (xr[c] - mu) * rstd * g[c] + bta[c];
        if (y)  y[row * D + c] = o;
        if (yb) yb[row * D + c] = (__bf16)o;
    }
}

// ---------------------------------------------------------------------------
extern "C" void kernel_launch(void* const* d_in, const int* in_sizes, int n_in,
                              void* d_out, int out_size, void* d_ws, size_t ws_size,
                              hipStream_t stream)
{
    (void)in_sizes; (void)n_in; (void)out_size; (void)ws_size;

    const float* tgt      = (const float*)d_in[0];
    const float* memory   = (const float*)d_in[1];
    const float* tgt_pos  = (const float*)d_in[2];
    const float* mem_pos  = (const float*)d_in[3];
    const int*   tgt_mask = (const int*)d_in[4];
    const int*   mem_mask = (const int*)d_in[5];
    const float* wsrc[10] = { (const float*)d_in[6],  (const float*)d_in[8],
                              (const float*)d_in[10], (const float*)d_in[12],
                              (const float*)d_in[14], (const float*)d_in[16],
                              (const float*)d_in[18], (const float*)d_in[20],
                              (const float*)d_in[28], (const float*)d_in[30] };
    const float *sa_bq=(const float*)d_in[7],  *sa_bk=(const float*)d_in[9];
    const float *sa_bv=(const float*)d_in[11], *sa_bo=(const float*)d_in[13];
    const float *ca_bq=(const float*)d_in[15], *ca_bk=(const float*)d_in[17];
    const float *ca_bv=(const float*)d_in[19], *ca_bo=(const float*)d_in[21];
    const float *ln1_g=(const float*)d_in[22], *ln1_b=(const float*)d_in[23];
    const float *ln2_g=(const float*)d_in[24], *ln2_b=(const float*)d_in[25];
    const float *ln3_g=(const float*)d_in[26], *ln3_b=(const float*)d_in[27];
    const float *bi1=(const float*)d_in[29],   *bi2=(const float*)d_in[31];

    const int Bn = 4, S = 1024, D = 512, Hf = 2048, NH = 8;
    const int R  = Bn * S;                     // 4096
    const long NT = (long)R * D;               // 2M elements

    // ---- workspace layout ----
    char* w = (char*)d_ws;
    auto alloc = [&](long bytes) { char* p = w; w += (bytes + 255) & ~255L; return p; };
    __bf16* wb[10];                            // bf16 weights
    const long wsz[10] = { (long)D*D, (long)D*D, (long)D*D, (long)D*D,
                           (long)D*D, (long)D*D, (long)D*D, (long)D*D,
                           (long)D*Hf, (long)Hf*D };
    for (int i = 0; i < 10; ++i) wb[i] = (__bf16*)alloc(wsz[i] * 2);
    __bf16* x1b  = (__bf16*)alloc(NT * 2);     // tgt+pos / later mq
    __bf16* tgtb = (__bf16*)alloc(NT * 2);     // tgt / later mk
    __bf16* memb = (__bf16*)alloc(NT * 2);     // memory (bf16)
    __bf16* qb   = (__bf16*)alloc(NT * 2);
    __bf16* kb   = (__bf16*)alloc(NT * 2);
    __bf16* vtb  = (__bf16*)alloc(NT * 2);     // per-head transposed V
    __bf16* aob  = (__bf16*)alloc(NT * 2);     // attention output
    __bf16* t2b  = (__bf16*)alloc(NT * 2);
    __bf16* hb   = (__bf16*)alloc((long)R * Hf * 2);
    float*  f0   = (float*)alloc(NT * 4);
    float*  f1   = (float*)alloc(NT * 4);      // t1
    float*  f2   = (float*)alloc(NT * 4);      // t2
    float*  out  = (float*)d_out;

    const dim3 blk(256);
    const dim3 gE((int)(NT / 4 + 255) / 256);
    const dim3 gLN(R);
    const dim3 gAttn(Bn * NH * (S / 16));
    const float scale = 0.125f;               // 1/sqrt(64)

    auto cvt = [&](const float* x, __bf16* y, long n) {
        cvt_bf16_kernel<<<dim3((int)(n / 4 + 255) / 256), blk, 0, stream>>>(
            x, y, (int)(n / 4));
    };

    // ---- one-time conversions (deterministic, every call) ----
    for (int i = 0; i < 10; ++i) cvt(wsrc[i], wb[i], wsz[i]);
    cvt(tgt, tgtb, NT);
    cvt(memory, memb, NT);

    // ---- self-attention ----
    add2_bf16_kernel<<<gE, blk, 0, stream>>>(tgt, tgt_pos, x1b, (int)(NT / 4));
    gemm_bf16_wmma<0><<<dim3(D/128, R/32), blk, 0, stream>>>(
        x1b, wb[0], sa_bq, nullptr, nullptr, qb, R, D, D);
    gemm_bf16_wmma<0><<<dim3(D/128, R/32), blk, 0, stream>>>(
        x1b, wb[1], sa_bk, nullptr, nullptr, kb, R, D, D);
    gemm_bf16_wmma<2><<<dim3(D/128, R/32), blk, 0, stream>>>(
        tgtb, wb[2], sa_bv, nullptr, nullptr, vtb, R, D, D);
    attn_wmma<<<gAttn, blk, 0, stream>>>(qb, kb, vtb, tgt_mask, aob,
                                         Bn, NH, S, S, scale);
    gemm_bf16_wmma<4><<<dim3(D/128, R/32), blk, 0, stream>>>(
        aob, wb[3], sa_bo, tgt, f0, nullptr, R, D, D);
    ln_kernel<<<gLN, blk, 0, stream>>>(f0, ln1_g, ln1_b, f1, nullptr, D);

    // ---- cross-attention ----
    add2_bf16_kernel<<<gE, blk, 0, stream>>>(f1, tgt_pos, x1b, (int)(NT / 4));
    add2_bf16_kernel<<<gE, blk, 0, stream>>>(memory, mem_pos, tgtb, (int)(NT / 4));
    gemm_bf16_wmma<0><<<dim3(D/128, R/32), blk, 0, stream>>>(
        x1b, wb[4], ca_bq, nullptr, nullptr, qb, R, D, D);
    gemm_bf16_wmma<0><<<dim3(D/128, R/32), blk, 0, stream>>>(
        tgtb, wb[5], ca_bk, nullptr, nullptr, kb, R, D, D);
    gemm_bf16_wmma<2><<<dim3(D/128, R/32), blk, 0, stream>>>(
        memb, wb[6], ca_bv, nullptr, nullptr, vtb, R, D, D);
    attn_wmma<<<gAttn, blk, 0, stream>>>(qb, kb, vtb, mem_mask, aob,
                                         Bn, NH, S, S, scale);
    gemm_bf16_wmma<4><<<dim3(D/128, R/32), blk, 0, stream>>>(
        aob, wb[7], ca_bo, f1, f0, nullptr, R, D, D);
    ln_kernel<<<gLN, blk, 0, stream>>>(f0, ln2_g, ln2_b, f2, t2b, D);

    // ---- FFN ----
    gemm_bf16_wmma<1><<<dim3(Hf/128, R/32), blk, 0, stream>>>(
        t2b, wb[8], bi1, nullptr, nullptr, hb, R, Hf, D);
    gemm_bf16_wmma<4><<<dim3(D/128, R/32), blk, 0, stream>>>(
        hb, wb[9], bi2, f2, f0, nullptr, R, D, Hf);
    ln_kernel<<<gLN, blk, 0, stream>>>(f0, ln3_g, ln3_b, out, nullptr, D);
}
